// APNet_mod_15023795601946
// MI455X (gfx1250) — matrix operand order repaired and checked by reading, not compiled, not run on hardware
//
#include <hip/hip_runtime.h>

typedef __attribute__((ext_vector_type(16))) _Float16 v16h;
typedef __attribute__((ext_vector_type(8)))  _Float16 v8h;
typedef __attribute__((ext_vector_type(8)))  float    v8f;

namespace {
constexpr int kB = 8, kA = 64, kN = 63, kP = 1953, kE = 5;
constexpr int kGR = 43, kGA = 21;
constexpr int OUT_RAD  = 0;
constexpr int OUT_AP   = kB * kA * kGR * kE;            // 110080
constexpr int OUT_DIST = OUT_AP + kB * kA * kGA * kE;   // 163840
constexpr int OUT_FD   = OUT_DIST + kB * kA * kN * 2;   // 228352
constexpr int ASTR_R = 72;                // radial A row stride (halves), 144B rows (16B aligned)
constexpr int ASTR_A = 136;               // angular A row stride (halves), 272B rows
constexpr int ABUF   = 32 * ASTR_A;       // 4352 halves; also covers radial 48*72=3456
constexpr int NCHUNK = (kP + 127) / 128;  // 16
}

__global__ __launch_bounds__(128)
void apnet_sf_kernel(const float* __restrict__ pos,    // [B,A,3]
                     const int*   __restrict__ Zt,     // [B,A]
                     const int*   __restrict__ nbh,    // [B,A,N]
                     const float* __restrict__ nmask,  // [B,A,N]
                     const float* __restrict__ cell,   // [B,3,3]
                     const float* __restrict__ coff,   // [B,A,N,3]
                     const float* __restrict__ pmask,  // [B,A,P]
                     const int*   __restrict__ ojj,    // [B,A,P] neighbor-slot j
                     const int*   __restrict__ okk,    // [B,A,P] neighbor-slot k
                     const float* __restrict__ zw,     // [10,5]
                     float* __restrict__ out)
{
  const int tid  = threadIdx.x;
  const int lane = tid & 31;
  const int wav  = tid >> 5;
  const int b  = blockIdx.x >> 6;       // blockIdx.x = b*64 + a
  const int a  = blockIdx.x & 63;
  const int ba = blockIdx.x;

  __shared__ float s_d[64];
  __shared__ float s_cut[64];
  __shared__ float s_nm[64];
  __shared__ float s_dv[64][4];
  __shared__ __align__(16) _Float16 s_z[64 * 8];     // raw one-hot element rows (f16)
  __shared__ __align__(16) _Float16 s_Am[ABUF];      // WMMA A staging (radial then angular)
  __shared__ __align__(16) _Float16 s_Bm[128 * 8];   // angular B rows (pairmask * one-hot)

  // -------- zero the A staging (covers all pad rows/cols for both phases) --------
  for (int i = tid; i < ABUF / 2; i += 128) ((unsigned*)s_Am)[i] = 0u;

  const float pax = pos[ba * 3 + 0];
  const float pay = pos[ba * 3 + 1];
  const float paz = pos[ba * 3 + 2];

  // -------- phase 1: per-neighbor distances, cutoffs, element one-hots, dists out --------
  if (tid < kN) {
    const int n = tid;
    const int j = nbh[ba * kN + n];
    const float* C = cell + b * 9;
    const float c0 = coff[(ba * kN + n) * 3 + 0];
    const float c1 = coff[(ba * kN + n) * 3 + 1];
    const float c2 = coff[(ba * kN + n) * 3 + 2];
    const float ox = c0 * C[0] + c1 * C[3] + c2 * C[6];
    const float oy = c0 * C[1] + c1 * C[4] + c2 * C[7];
    const float oz = c0 * C[2] + c1 * C[5] + c2 * C[8];
    const int jg = b * kA + j;
    const float dx = pos[jg * 3 + 0] + ox - pax;
    const float dy = pos[jg * 3 + 1] + oy - pay;
    const float dz = pos[jg * 3 + 2] + oz - paz;
    const float d2 = dx * dx + dy * dy + dz * dz;
    const float m  = nmask[ba * kN + n];
    const float dd = (m > 0.f) ? __builtin_sqrtf((m > 0.f) ? d2 : 1.f) : 0.f;
    s_d[n] = dd;
    s_dv[n][0] = dx; s_dv[n][1] = dy; s_dv[n][2] = dz; s_dv[n][3] = 0.f;
    s_nm[n] = m;
    // cosine cutoff: 0.5*(cos(pi*d/8)+1)*(d<8)
    s_cut[n] = (dd < 8.f) ? 0.5f * (__cosf(0.39269908169872414f * dd) + 1.f) : 0.f;
    // element one-hot row (Behler embedding) as f16
    const int zv = Zt[jg];
    #pragma unroll
    for (int e = 0; e < kE; ++e) s_z[n * 8 + e] = (_Float16)zw[zv * kE + e];
    s_z[n * 8 + 5] = (_Float16)0.f;
    s_z[n * 8 + 6] = (_Float16)0.f;
    s_z[n * 8 + 7] = (_Float16)0.f;
    // stacked distance / inverse-distance output
    out[OUT_DIST + (ba * kN + n) * 2 + 0] = dd;
    out[OUT_DIST + (ba * kN + n) * 2 + 1] = 1.f / dd;
  }
  __syncthreads();

  // -------- Fermi-Dirac reaction/product coordinate (pair_dist[b,1], pair_dist[b,2]) --------
  if (a == 0 && tid == 0) {
    const float d1 = s_d[1], d2 = s_d[2];
    const float f1 = 1.f / (1.f + __expf(10.f * (d1 - 1.5f)));
    const float f2 = 1.f / (1.f + __expf(10.f * (d2 - 1.5f)));
    out[OUT_FD + b] = f1 * f2;
  }

  // -------- phase 2: radial SF as (43x63)@(63x5) f16 WMMA GEMM --------
  // Fill A[g][n] = exp(-100*(d-off_g)^2) * fc(d), row-major, stride 72 halves.
  for (int i = tid; i < kGR * kN; i += 128) {
    const int g = i / kN;
    const int n = i - g * kN;
    const float x = s_d[n] - (0.8f + 0.1f * (float)g);
    s_Am[g * ASTR_R + n] = (_Float16)(__expf(-100.f * x * x) * s_cut[n]);
  }
  __syncthreads();

  if (wav < 3) {                 // wave-uniform: EXEC all-ones inside WMMA region
    const int lr = lane & 15;
    const int hi = lane >> 4;
    const int row = wav * 16 + lr;
    v8f acc = {};
    #pragma unroll
    for (int kt = 0; kt < 2; ++kt) {
      // A operand: lane's two K-octets (ISA 16-bit A 16x32 layout)
      const _Float16* rp = s_Am + row * ASTR_R + kt * 32 + hi * 8;
      const v8h a0 = *(const v8h*)rp;
      const v8h a1 = *(const v8h*)(rp + 16);
      v16h Af;
      #pragma unroll
      for (int q = 0; q < 8; ++q) { Af[q] = a0[q]; Af[q + 8] = a1[q]; }
      // B operand: lane holds column N = lane&15, K in table order -> masked one-hot
      v16h Bf = {};
      const int bn = lane & 15;
      #pragma unroll
      for (int s = 0; s < 16; ++s) {
        const int Kg = kt * 32 + (s < 8 ? hi * 8 + s : 16 + hi * 8 + (s - 8));
        if (bn < kE && Kg < kN) Bf[s] = s_z[Kg * 8 + bn] * (_Float16)s_nm[Kg];
      }
      acc = __builtin_amdgcn_wmma_f32_16x16x32_f16(false, Af, false, Bf,
                                                   (short)0, acc, false, false);
    }
    // D layout: lane<16 -> N=lane (e), M=r ; lane>=16 -> N=lane-16, M=8+r
    const int e = lane & 15;
    if (e < kE) {
      #pragma unroll
      for (int r = 0; r < 8; ++r) {
        const int g = wav * 16 + hi * 8 + r;
        if (g < kGR) out[OUT_RAD + ba * (kGR * kE) + g * kE + e] = acc[r];
      }
    }
  }
  __syncthreads();

  // -------- phase 3: angular SF as (21x1953)@(1953x5) WMMA GEMM, K streamed --------
  // Re-zero A staging so pad rows (g>=21) / pad cols stay zero; rows 0..20 are
  // fully rewritten every chunk.
  for (int i = tid; i < ABUF / 2; i += 128) ((unsigned*)s_Am)[i] = 0u;
  __syncthreads();

  v8f accA = {};
  const int pb = ba * kP;
  for (int ch = 0; ch < NCHUNK; ++ch) {
    const int p = ch * 128 + tid;
    if (p < kP) {
      const int jj = ojj[pb + p];
      const int kk = okk[pb + p];
      const float pm  = pmask[pb + p];
      const float rij = s_d[jj];
      const float rik = s_d[kk];
      const float ex = s_dv[kk][0] - s_dv[jj][0];
      const float ey = s_dv[kk][1] - s_dv[jj][1];
      const float ez = s_dv[kk][2] - s_dv[jj][2];
      const float rjk2 = ex * ex + ey * ey + ez * ez;
      const float den = 2.f * rij * rik;
      const float ct = (pm > 0.f)
          ? (rij * rij + rik * rik - rjk2) / ((den > 0.f) ? den : 1.f) : 0.f;
      const float cut = s_cut[jj] * s_cut[kk];
      #pragma unroll
      for (int g = 0; g < kGA; ++g) {
        const float x = ct - (-1.f + 0.1f * (float)g);
        s_Am[g * ASTR_A + tid] = (_Float16)(__expf(-25.f * x * x) * cut);
      }
      const _Float16 pmh = (_Float16)pm;
      #pragma unroll
      for (int e = 0; e < 8; ++e) s_Bm[tid * 8 + e] = s_z[kk * 8 + e] * pmh;
    } else {
      #pragma unroll
      for (int g = 0; g < kGA; ++g) s_Am[g * ASTR_A + tid] = (_Float16)0.f;
      #pragma unroll
      for (int e = 0; e < 8; ++e) s_Bm[tid * 8 + e] = (_Float16)0.f;
    }
    __syncthreads();

    if (wav < 2) {               // wave-uniform
      const int lr = lane & 15;
      const int hi = lane >> 4;
      const int row = wav * 16 + lr;
      #pragma unroll
      for (int kt = 0; kt < 4; ++kt) {
        const _Float16* rp = s_Am + row * ASTR_A + kt * 32 + hi * 8;
        const v8h a0 = *(const v8h*)rp;
        const v8h a1 = *(const v8h*)(rp + 16);
        v16h Af;
        #pragma unroll
        for (int q = 0; q < 8; ++q) { Af[q] = a0[q]; Af[q + 8] = a1[q]; }
        const int bn = lane & 15;
        v16h Bf = {};
        #pragma unroll
        for (int s = 0; s < 16; ++s) if (bn < kE) Bf[s] = s_Bm[(kt * 32 + (s < 8 ? hi * 8 + s : 16 + hi * 8 + (s - 8))) * 8 + bn];
        accA = __builtin_amdgcn_wmma_f32_16x16x32_f16(false, Af, false, Bf,
                                                      (short)0, accA, false, false);
      }
    }
    __syncthreads();
  }

  if (wav < 2) {
    const int hi = lane >> 4;
    const int e = lane & 15;
    if (e < kE) {
      #pragma unroll
      for (int r = 0; r < 8; ++r) {
        const int g = wav * 16 + hi * 8 + r;
        if (g < kGA) out[OUT_AP + ba * (kGA * kE) + g * kE + e] = accA[r];
      }
    }
  }
}

extern "C" void kernel_launch(void* const* d_in, const int* in_sizes, int n_in,
                              void* d_out, int out_size, void* d_ws, size_t ws_size,
                              hipStream_t stream) {
  (void)in_sizes; (void)n_in; (void)out_size; (void)d_ws; (void)ws_size;
  const float* pos   = (const float*)d_in[0];
  const int*   Zt    = (const int*)  d_in[1];
  const int*   nbh   = (const int*)  d_in[2];
  const float* nmask = (const float*)d_in[3];
  const float* cell  = (const float*)d_in[4];
  const float* coff  = (const float*)d_in[5];
  // d_in[6]=idx_j, d_in[7]=idx_k are redundant given slot indices + cached dvec
  const float* pmask = (const float*)d_in[8];
  const int*   ojj   = (const int*)  d_in[9];
  const int*   okk   = (const int*)  d_in[10];
  const float* zw    = (const float*)d_in[11];
  float* out = (float*)d_out;

  apnet_sf_kernel<<<dim3(kB * kA), dim3(128), 0, stream>>>(
      pos, Zt, nbh, nmask, cell, coff, pmask, ojj, okk, zw, out);
}
